// EnhancedGNNTransformerEncoder_49752901157009
// MI455X (gfx1250) — matrix operand order repaired and checked by reading, not compiled, not run on hardware
//
#include <hip/hip_runtime.h>
#include <stdint.h>

// ---------------- problem constants ----------------
constexpr int NN  = 50000;
constexpr int EE  = 800000;
constexpr int INF_DIM = 128;
constexpr int HH  = 8;
constexpr int CC  = 32;
constexpr int HC  = 256;
constexpr int GG  = 64;
constexpr int OUT = 64;
constexpr int NC  = 1024;          // fused q|k|v|s output width
constexpr int MTILES = NN / 16;    // 3125 (exact)

// ---------------- types ----------------
typedef __attribute__((ext_vector_type(16))) __bf16   v16bf;
typedef __attribute__((ext_vector_type(8)))  float    v8f;
typedef __attribute__((ext_vector_type(4)))  unsigned v4u;
typedef __attribute__((ext_vector_type(4)))  float    v4f;

union FragAB { v16bf v; v4u q[2]; unsigned u[8]; };
union FragC  { v8f  v; float f[8]; };

__device__ __forceinline__ unsigned short f32_to_bf16(float f) {
    union { float f; unsigned u; } x; x.f = f;
    unsigned r = x.u + 0x7FFFu + ((x.u >> 16) & 1u);   // round-to-nearest-even
    return (unsigned short)(r >> 16);
}

__device__ __forceinline__ void atomicMaxF(float* addr, float val) {
    if (val >= 0.0f) atomicMax((int*)addr, __float_as_int(val));
    else             atomicMin((unsigned*)addr, __float_as_uint(val));
}

// ---------------- prep kernels ----------------
__global__ void f32_to_bf16_kernel(const float* __restrict__ src,
                                   unsigned short* __restrict__ dst, int n) {
    int gid = blockIdx.x * blockDim.x + threadIdx.x;
    if (gid < n) dst[gid] = f32_to_bf16(src[gid]);
}

// Pack W{q,k,v,s} [K,256] fp32 into WMMA B-fragment order for B = [K, 1024] bf16.
// Storage: tile t = kt*64+nt; Bp[t*256 + lane*8 + j] holds K-pair (k0,k0+1) for
// column n = nt*16 + (lane&15), k0 = kt*32 + (lane>=16 ? 8 : 0) + (j>=4 ? 16 : 0) + 2*(j&3).
__global__ void pack_weights_kernel(const float* __restrict__ Wq,
                                    const float* __restrict__ Wk,
                                    const float* __restrict__ Wv,
                                    const float* __restrict__ Ws,
                                    int K, unsigned* __restrict__ pack) {
    int gid = blockIdx.x * blockDim.x + threadIdx.x;
    int total = (K / 32) * 64 * 32 * 8;
    if (gid >= total) return;
    int j    = gid & 7;
    int lane = (gid >> 3) & 31;
    int tile = gid >> 8;
    int nt   = tile & 63;
    int kt   = tile >> 6;
    int n    = nt * 16 + (lane & 15);
    int k0   = kt * 32 + ((lane >> 4) << 3) + ((j >> 2) << 4) + 2 * (j & 3);
    int sel  = n >> 8;
    int col  = n & 255;
    const float* W = (sel == 0) ? Wq : (sel == 1) ? Wk : (sel == 2) ? Wv : Ws;
    unsigned short lo = f32_to_bf16(W[(size_t)k0       * HC + col]);
    unsigned short hi = f32_to_bf16(W[(size_t)(k0 + 1) * HC + col]);
    pack[gid] = (unsigned)lo | ((unsigned)hi << 16);
}

__global__ void pack_bias_kernel(const float* __restrict__ bq, const float* __restrict__ bk,
                                 const float* __restrict__ bv, const float* __restrict__ bs,
                                 float* __restrict__ out) {
    int n = blockIdx.x * blockDim.x + threadIdx.x;
    if (n >= NC) return;
    int sel = n >> 8, col = n & 255;
    const float* b = (sel == 0) ? bq : (sel == 1) ? bk : (sel == 2) ? bv : bs;
    out[n] = b[col];
}

// ---------------- fused q|k|v|s GEMM via bf16 WMMA ----------------
// C[N,1024] = A[N,K](bf16) @ B[K,1024](bf16, prepacked) + bias.
// Block = 256 threads = 8 waves; wave -> one 16-row tile x four 16-col tiles.
__global__ __launch_bounds__(256)
void gemm_qkvs_wmma(const unsigned short* __restrict__ A, int K,
                    const unsigned* __restrict__ Bp,
                    const float* __restrict__ bias,
                    float* __restrict__ Cout) {
    const int lane  = threadIdx.x & 31;
    const int wave  = threadIdx.x >> 5;
    const int mTile = blockIdx.x * 8 + wave;
    if (mTile >= MTILES) return;                 // wave-uniform: EXEC stays full
    const int nBase = blockIdx.y * 64;
    const int half  = lane >> 4;                 // 0|1
    const int l16   = lane & 15;

    // A fragment source: row = mTile*16 + l16; lanes 0-15 need K 0..7 & 16..23,
    // lanes 16-31 need K 8..15 & 24..31 (16-bit A layout, ISA 7.12.2).
    const unsigned short* Arow = A + (size_t)(mTile * 16 + l16) * K + half * 8;

    FragC acc[4];
    #pragma unroll
    for (int t = 0; t < 4; ++t)
        #pragma unroll
        for (int i = 0; i < 8; ++i) acc[t].f[i] = 0.0f;

    const int kTiles = K >> 5;
    const int ntBase = nBase >> 4;
    for (int kt = 0; kt < kTiles; ++kt) {
        FragAB a;
        const unsigned short* ap = Arow + kt * 32;
        a.q[0] = *reinterpret_cast<const v4u*>(ap);        // K off..off+7
        a.q[1] = *reinterpret_cast<const v4u*>(ap + 16);   // K off+16..off+23
        const unsigned* bbase = Bp + (size_t)(kt * 64) * 256 + lane * 8;
        #pragma unroll
        for (int t = 0; t < 4; ++t) {
            FragAB b;
            const unsigned* bp = bbase + (size_t)(ntBase + t) * 256;
            b.q[0] = *reinterpret_cast<const v4u*>(bp);
            b.q[1] = *reinterpret_cast<const v4u*>(bp + 4);
            acc[t].v = __builtin_amdgcn_wmma_f32_16x16x32_bf16(
                false, a.v, false, b.v, (short)0, acc[t].v, false, false);
        }
    }

    // C/D layout: lane = 16*half + col, VGPR r -> row = 8*half + r.
    const int rowBase = mTile * 16 + half * 8;
    #pragma unroll
    for (int t = 0; t < 4; ++t) {
        const int col = nBase + t * 16 + l16;
        const float bb = bias[col];
        float* cp = Cout + (size_t)rowBase * NC + col;
        #pragma unroll
        for (int r = 0; r < 8; ++r)
            cp[(size_t)r * NC] = acc[t].f[r] + bb;
    }
}

// ---------------- attention (edge) kernels ----------------
__global__ void attn_init(float* __restrict__ agg, float* __restrict__ m,
                          float* __restrict__ ssum) {
    int gid = blockIdx.x * blockDim.x + threadIdx.x;   // over NN*256
    agg[gid] = 0.0f;
    if (gid < NN * HH) { m[gid] = -__builtin_inff(); ssum[gid] = 0.0f; }
}

__global__ void edge_logits_max(const int* __restrict__ src, const int* __restrict__ dst,
                                const float* __restrict__ qkvs,
                                float* __restrict__ logits, float* __restrict__ m) {
    int gid = blockIdx.x * blockDim.x + threadIdx.x;   // over EE*HH
    int e = gid >> 3, h = gid & 7;
    int s = src[e], d = dst[e];
    const v4f* qp = reinterpret_cast<const v4f*>(qkvs + (size_t)d * NC + h * CC);       // q
    const v4f* kp = reinterpret_cast<const v4f*>(qkvs + (size_t)s * NC + 256 + h * CC); // k
    float acc = 0.0f;
    #pragma unroll
    for (int i = 0; i < 8; ++i) {
        v4f q4 = qp[i], k4 = kp[i];
        acc += q4.x * k4.x + q4.y * k4.y + q4.z * k4.z + q4.w * k4.w;
    }
    float logit = acc * 0.17677669529663687f;          // 1/sqrt(32)
    logits[gid] = logit;
    atomicMaxF(m + (size_t)d * HH + h, logit);
}

__global__ void edge_exp_sum(const int* __restrict__ dst, float* __restrict__ logits,
                             const float* __restrict__ m, float* __restrict__ ssum) {
    int gid = blockIdx.x * blockDim.x + threadIdx.x;
    int e = gid >> 3, h = gid & 7;
    int d = dst[e];
    float p = __expf(logits[gid] - m[(size_t)d * HH + h]);
    logits[gid] = p;
    atomicAdd(ssum + (size_t)d * HH + h, p);
}

__global__ void edge_aggregate(const int* __restrict__ src, const int* __restrict__ dst,
                               const float* __restrict__ logits, const float* __restrict__ ssum,
                               const float* __restrict__ qkvs, float* __restrict__ agg) {
    int gid = blockIdx.x * blockDim.x + threadIdx.x;
    int e = gid >> 3, h = gid & 7;
    int s = src[e], d = dst[e];
    float alpha = logits[gid] / (ssum[(size_t)d * HH + h] + 1e-16f);
    const float* vp = qkvs + (size_t)s * NC + 512 + h * CC;   // v
    float* op = agg + (size_t)d * HC + h * CC;
    #pragma unroll
    for (int c = 0; c < CC; ++c) atomicAdd(op + c, alpha * vp[c]);
}

__global__ void finalize_layer(float* __restrict__ agg, const float* __restrict__ qkvs,
                               unsigned short* __restrict__ bfout) {
    int gid = blockIdx.x * blockDim.x + threadIdx.x;   // over NN*256
    int n = gid >> 8, c = gid & 255;
    float v = agg[gid] + qkvs[(size_t)n * NC + 768 + c];  // + root skip (s)
    v = fmaxf(v, 0.0f);                                    // ReLU
    agg[gid]   = v;                                        // fp32 h (for pooling)
    bfout[gid] = f32_to_bf16(v);                           // bf16 h (next GEMM)
}

// ---------------- pooling + final FC ----------------
__global__ void pool_zero(float* __restrict__ pooled, float* __restrict__ counts) {
    int gid = blockIdx.x * blockDim.x + threadIdx.x;
    if (gid < GG * HC) pooled[gid] = 0.0f;
    if (gid < GG) counts[gid] = 0.0f;
}

__global__ void pool_count(const int* __restrict__ batch, float* __restrict__ counts) {
    int gid = blockIdx.x * blockDim.x + threadIdx.x;
    if (gid < NN) atomicAdd(counts + batch[gid], 1.0f);
}

__global__ void pool_sum(const int* __restrict__ batch, const float* __restrict__ h,
                         float* __restrict__ pooled) {
    int gid = blockIdx.x * blockDim.x + threadIdx.x;   // over NN*256
    int n = gid >> 8, c = gid & 255;
    atomicAdd(pooled + (size_t)batch[n] * HC + c, h[gid]);
}

__global__ void final_fc(const float* __restrict__ pooled, const float* __restrict__ counts,
                         const float* __restrict__ Wfc, const float* __restrict__ bfc,
                         float* __restrict__ out) {
    int g = blockIdx.x, o = threadIdx.x;               // 64 x 64
    float cnt = fmaxf(counts[g], 1.0f);
    float acc = 0.0f;
    for (int i = 0; i < HC; ++i) acc += pooled[(size_t)g * HC + i] * Wfc[(size_t)i * OUT + o];
    out[(size_t)g * OUT + o] = acc / cnt + bfc[o];
}

// ---------------- host driver ----------------
extern "C" void kernel_launch(void* const* d_in, const int* in_sizes, int n_in,
                              void* d_out, int out_size, void* d_ws, size_t ws_size,
                              hipStream_t stream) {
    const float* x     = (const float*)d_in[0];
    const int*   ei    = (const int*)d_in[1];
    const int*   srcI  = ei;
    const int*   dstI  = ei + EE;
    const int*   batch = (const int*)d_in[2];
    const float* Wq0 = (const float*)d_in[3];  const float* bq0 = (const float*)d_in[4];
    const float* Wk0 = (const float*)d_in[5];  const float* bk0 = (const float*)d_in[6];
    const float* Wv0 = (const float*)d_in[7];  const float* bv0 = (const float*)d_in[8];
    const float* Ws0 = (const float*)d_in[9];  const float* bs0 = (const float*)d_in[10];
    const float* WqL = (const float*)d_in[11]; const float* bqL = (const float*)d_in[12];
    const float* WkL = (const float*)d_in[13]; const float* bkL = (const float*)d_in[14];
    const float* WvL = (const float*)d_in[15]; const float* bvL = (const float*)d_in[16];
    const float* WsL = (const float*)d_in[17]; const float* bsL = (const float*)d_in[18];
    const float* Wfc = (const float*)d_in[19]; const float* bfc = (const float*)d_in[20];

    // ---- carve workspace (~312 MB) ----
    char* p = (char*)d_ws;
    auto carve = [&](size_t bytes) -> void* {
        void* r = (void*)p;
        p += (bytes + 255) & ~(size_t)255;
        return r;
    };
    unsigned short* bf_in  = (unsigned short*)carve((size_t)NN * HC * 2);   // bf16 activations
    float*          qkvs   = (float*)carve((size_t)NN * NC * 4);            // fused q|k|v|s
    float*          agg    = (float*)carve((size_t)NN * HC * 4);            // attn accum / h
    float*          logits = (float*)carve((size_t)EE * HH * 4);
    float*          mbuf   = (float*)carve((size_t)NN * HH * 4);
    float*          ssum   = (float*)carve((size_t)NN * HH * 4);
    unsigned*       pack   = (unsigned*)carve((size_t)(256 + 3 * 512) * 256 * 4);
    float*          biasB  = (float*)carve((size_t)4 * NC * 4);
    float*          pooled = (float*)carve((size_t)GG * HC * 4);
    float*          counts = (float*)carve((size_t)GG * 4);

    const size_t pack0 = 0;                 // layer 0: 256 tiles
    const size_t packL = 256 * 256;         // 65536 uints, then 131072 per later layer

    // ---- pack weights + biases (once per call) ----
    pack_weights_kernel<<<65536 / 256, 256, 0, stream>>>(Wq0, Wk0, Wv0, Ws0, INF_DIM, pack + pack0);
    for (int l = 0; l < 3; ++l) {
        size_t wo = (size_t)l * HC * HC;
        pack_weights_kernel<<<131072 / 256, 256, 0, stream>>>(
            WqL + wo, WkL + wo, WvL + wo, WsL + wo, HC,
            pack + packL + (size_t)l * 512 * 256);
    }
    pack_bias_kernel<<<4, 256, 0, stream>>>(bq0, bk0, bv0, bs0, biasB);
    for (int l = 0; l < 3; ++l)
        pack_bias_kernel<<<4, 256, 0, stream>>>(bqL + l * HC, bkL + l * HC,
                                                bvL + l * HC, bsL + l * HC,
                                                biasB + (size_t)(l + 1) * NC);

    // ---- layer 0 input: x -> bf16 ----
    f32_to_bf16_kernel<<<(NN * INF_DIM + 255) / 256, 256, 0, stream>>>(x, bf_in, NN * INF_DIM);

    const dim3 gemmGrid((MTILES + 7) / 8, NC / 64, 1);   // 391 x 16
    const int edgeBlocks = (EE * HH) / 256;              // 25000
    const int nodeBlocks = (NN * HC) / 256;              // 50000

    for (int layer = 0; layer < 4; ++layer) {
        const int K = (layer == 0) ? INF_DIM : HC;
        const unsigned* pw = pack + ((layer == 0) ? pack0
                                                  : packL + (size_t)(layer - 1) * 512 * 256);
        const float* pb = biasB + (size_t)layer * NC;

        gemm_qkvs_wmma<<<gemmGrid, 256, 0, stream>>>(bf_in, K, pw, pb, qkvs);
        attn_init<<<nodeBlocks, 256, 0, stream>>>(agg, mbuf, ssum);
        edge_logits_max<<<edgeBlocks, 256, 0, stream>>>(srcI, dstI, qkvs, logits, mbuf);
        edge_exp_sum<<<edgeBlocks, 256, 0, stream>>>(dstI, logits, mbuf, ssum);
        edge_aggregate<<<edgeBlocks, 256, 0, stream>>>(srcI, dstI, logits, ssum, qkvs, agg);
        finalize_layer<<<nodeBlocks, 256, 0, stream>>>(agg, qkvs, bf_in);
    }

    // ---- global mean pool + final linear ----
    pool_zero<<<(GG * HC + 255) / 256 + 1, 256, 0, stream>>>(pooled, counts);
    pool_count<<<(NN + 255) / 256, 256, 0, stream>>>(batch, counts);
    pool_sum<<<nodeBlocks, 256, 0, stream>>>(batch, agg, pooled);
    final_fc<<<GG, OUT, 0, stream>>>(pooled, counts, Wfc, bfc, (float*)d_out);
}
